// GNN_LCG_15839839387885
// MI455X (gfx1250) — matrix operand order repaired and checked by reading, not compiled, not run on hardware
//
#include <hip/hip_runtime.h>
#include <hip/hip_bf16.h>
#include <math.h>

// ---------------------------------------------------------------------------
// CDNA5 (gfx1250) bf16 WMMA GNN implementation.
//   D = 128 feature dim everywhere; wave32; v_wmma_f32_16x16x32_bf16.
// ---------------------------------------------------------------------------

typedef __bf16 bf16_t;
typedef __attribute__((ext_vector_type(16))) bf16_t v16bf;
typedef __attribute__((ext_vector_type(8)))  float  v8f;

#define WG 256
#define D  128

// ---- WMMA fragment index helpers (per CDNA5 ISA 7.12.2, 16-bit layouts) ----
// A (16x32, MxK): lane = half*16 + m ; vgpr v holds K = kt*32 + a_kidx(v,half), +1
__device__ __forceinline__ int a_kidx(int v, int half) {
    return 2 * (v & 3) + 8 * half + 16 * (v >> 2);
}

// B fragments are pre-swizzled into contiguous per-lane 32B blocks:
// frag block (kt,nt): 32 lanes x 16 bf16.
__device__ __forceinline__ v16bf load_bfrag(const bf16_t* __restrict__ Wf,
                                            int kt, int nt, int lane) {
    return *(const v16bf*)(Wf + (((size_t)(kt * 8 + nt) * 32 + lane) << 4));
}

__device__ __forceinline__ v8f wmma_bf16(v16bf a, v16bf b, v8f c) {
    return __builtin_amdgcn_wmma_f32_16x16x32_bf16(false, a, false, b,
                                                   (short)0, c, false, false);
}

// Monotonic float<->uint mapping for atomicMax-based segment max.
__device__ __forceinline__ unsigned f2ord(float f) {
    unsigned u = __float_as_uint(f);
    return (u & 0x80000000u) ? ~u : (u | 0x80000000u);
}
__device__ __forceinline__ float ord2f(unsigned u) {
    return (u & 0x80000000u) ? __uint_as_float(u & 0x7fffffffu)
                             : __uint_as_float(~u);
}

// ---------------------------------------------------------------------------
// Weight pre-swizzle: f32 row-major [K x 128]  ->  bf16 B-fragment order.
// out[(((kt*8+nt)*32+lane)*16 + 2v+p)] = W[(kt*32 + a_kidx(v,half) + p)*128 + nt*16 + n]
// ---------------------------------------------------------------------------
__global__ void swizzle_w_kernel(const float* __restrict__ W, int Ktiles,
                                 bf16_t* __restrict__ out) {
    const int idx = blockIdx.x * WG + threadIdx.x;
    if (idx >= Ktiles * 4096) return;
    const int e    = idx & 15;
    const int lane = (idx >> 4) & 31;
    const int nt   = (idx >> 9) & 7;
    const int kt   = idx >> 12;
    const int v = e >> 1, p = e & 1;
    const int half = lane >> 4, n = lane & 15;
    const int k = kt * 32 + a_kidx(v, half) + p;
    out[idx] = (bf16_t)W[(size_t)k * D + nt * 16 + n];
}

// ---------------------------------------------------------------------------
// Fused 2-layer MLP:  Y = relu(X @ W1 + b1) @ W2 + b2        (K = N_out = 128)
// One block = 8 waves = 128 rows. Each wave: 16-row strip x 128 cols.
// Layer-1 result goes through a per-wave LDS bf16 tile (stride 132 to
// de-conflict banks) and is re-read in A-fragment layout for layer 2.
// xorPair!=0 reads input row r^1 (literal <-> negated-literal swap).
// ---------------------------------------------------------------------------
__global__ __launch_bounds__(WG)
void mlp2_fused_kernel(const float* __restrict__ X, int N, int xorPair,
                       const bf16_t* __restrict__ W1f, const float* __restrict__ b1,
                       const bf16_t* __restrict__ W2f, const float* __restrict__ b2,
                       float* __restrict__ Y) {
    __shared__ bf16_t Hs[8][16 * 132];
    const int wave = threadIdx.x >> 5, lane = threadIdx.x & 31;
    const int m = lane & 15, half = lane >> 4;
    const int rowBase = blockIdx.x * 128 + wave * 16;

    int arow = rowBase + m;
    if (xorPair) arow ^= 1;
    if (arow >= N) arow = 0;                 // clamp; stores are masked
    const float* __restrict__ xrow = X + (size_t)arow * D;

    v8f acc[8] = {};
#pragma unroll
    for (int kt = 0; kt < 4; ++kt) {
        v16bf a;
#pragma unroll
        for (int v = 0; v < 8; ++v) {
            const int k = kt * 32 + a_kidx(v, half);
            const float2 f = *(const float2*)(xrow + k);
            a[2 * v]     = (bf16_t)f.x;
            a[2 * v + 1] = (bf16_t)f.y;
        }
#pragma unroll
        for (int nt = 0; nt < 8; ++nt)
            acc[nt] = wmma_bf16(a, load_bfrag(W1f, kt, nt, lane), acc[nt]);
    }

    // bias + ReLU -> per-wave LDS tile (C/D layout: row v+8*half, col nt*16+m)
#pragma unroll
    for (int nt = 0; nt < 8; ++nt)
#pragma unroll
        for (int v = 0; v < 8; ++v) {
            const int col = nt * 16 + m;
            float h = acc[nt][v] + b1[col];
            h = h > 0.0f ? h : 0.0f;
            Hs[wave][(v + 8 * half) * 132 + col] = (bf16_t)h;
        }
    __syncthreads();

    v8f acc2[8] = {};
#pragma unroll
    for (int kt = 0; kt < 4; ++kt) {
        v16bf a;
#pragma unroll
        for (int v = 0; v < 8; ++v) {
            const int k = kt * 32 + a_kidx(v, half);
            a[2 * v]     = Hs[wave][m * 132 + k];
            a[2 * v + 1] = Hs[wave][m * 132 + k + 1];
        }
#pragma unroll
        for (int nt = 0; nt < 8; ++nt)
            acc2[nt] = wmma_bf16(a, load_bfrag(W2f, kt, nt, lane), acc2[nt]);
    }

#pragma unroll
    for (int nt = 0; nt < 8; ++nt)
#pragma unroll
        for (int v = 0; v < 8; ++v) {
            const int r = rowBase + v + 8 * half;
            if (r < N) {
                const int col = nt * 16 + m;
                Y[(size_t)r * D + col] = acc2[nt][v] + b2[col];
            }
        }
}

// ---------------------------------------------------------------------------
// Single-layer update GEMM with virtual concat of up to 3 [N x 128] sources:
//   Y = [S0 | S1 | S2] @ W + b,  K = Ktiles*32 (256 or 384).
// ---------------------------------------------------------------------------
__global__ __launch_bounds__(WG)
void update_gemm_kernel(const float* __restrict__ S0, const float* __restrict__ S1,
                        const float* __restrict__ S2, int N,
                        const bf16_t* __restrict__ Wf, const float* __restrict__ bias,
                        int Ktiles, float* __restrict__ Y) {
    const int wave = threadIdx.x >> 5, lane = threadIdx.x & 31;
    const int m = lane & 15, half = lane >> 4;
    const int rowBase = blockIdx.x * 128 + wave * 16;
    int arow = rowBase + m;
    if (arow >= N) arow = 0;
    const float* __restrict__ p0 = S0 + (size_t)arow * D;
    const float* __restrict__ p1 = S1 + (size_t)arow * D;
    const float* __restrict__ p2 = S2 ? (S2 + (size_t)arow * D) : p0;

    v8f acc[8] = {};
    for (int kt = 0; kt < Ktiles; ++kt) {
        v16bf a;
#pragma unroll
        for (int v = 0; v < 8; ++v) {
            const int k = kt * 32 + a_kidx(v, half);
            const int s = k >> 7;  // which concat source
            const float* sp = (s == 0) ? p0 : ((s == 1) ? p1 : p2);
            const float2 f = *(const float2*)(sp + (k & 127));
            a[2 * v]     = (bf16_t)f.x;
            a[2 * v + 1] = (bf16_t)f.y;
        }
#pragma unroll
        for (int nt = 0; nt < 8; ++nt)
            acc[nt] = wmma_bf16(a, load_bfrag(Wf, kt, nt, lane), acc[nt]);
    }

#pragma unroll
    for (int nt = 0; nt < 8; ++nt)
#pragma unroll
        for (int v = 0; v < 8; ++v) {
            const int r = rowBase + v + 8 * half;
            if (r < N) {
                const int col = nt * 16 + m;
                Y[(size_t)r * D + col] = acc[nt][v] + bias[col];
            }
        }
}

// ---------------------------------------------------------------------------
// Per-node attention scalar:  out[n] = dot(X[n,:], att[:128])   (1 wave / row)
// ---------------------------------------------------------------------------
__global__ __launch_bounds__(WG)
void nodedot_kernel(const float* __restrict__ X, int N,
                    const float* __restrict__ att, float* __restrict__ out) {
    const int wave = threadIdx.x >> 5, lane = threadIdx.x & 31;
    const int row = blockIdx.x * 8 + wave;
    if (row >= N) return;
    const float* x = X + (size_t)row * D;
    float s = 0.0f;
#pragma unroll
    for (int i = 0; i < 4; ++i) s += x[lane + 32 * i] * att[lane + 32 * i];
#pragma unroll
    for (int off = 16; off > 0; off >>= 1) s += __shfl_xor(s, off, 32);
    if (lane == 0) out[row] = s;
}

__global__ void seginit_kernel(unsigned* mA, float* sA, unsigned* mB, float* sB, int n) {
    int i = blockIdx.x * WG + threadIdx.x;
    if (i >= n) return;
    mA[i] = 0u; sA[i] = 0.0f; mB[i] = 0u; sB[i] = 0.0f;
}

// w[e] = leaky_relu(sA[iA[e]] + sB[iB[e]]); segment max via ordinal atomicMax.
__global__ void edge_w_kernel(const int* __restrict__ iA, const float* __restrict__ sA,
                              const int* __restrict__ iB, const float* __restrict__ sB,
                              const int* __restrict__ seg, unsigned* __restrict__ segmax,
                              float* __restrict__ w, int E_) {
    int e = blockIdx.x * WG + threadIdx.x;
    if (e >= E_) return;
    float v = sA[iA[e]] + sB[iB[e]];
    v = v > 0.0f ? v : 0.2f * v;
    w[e] = v;
    atomicMax(&segmax[seg[e]], f2ord(v));
}

// w[e] <- exp(w[e] - max[seg]); segment sum of exps.
__global__ void edge_exp_kernel(const int* __restrict__ seg,
                                const unsigned* __restrict__ segmax,
                                float* __restrict__ w, float* __restrict__ segsum,
                                int E_) {
    int e = blockIdx.x * WG + threadIdx.x;
    if (e >= E_) return;
    float ex = expf(w[e] - ord2f(segmax[seg[e]]));
    w[e] = ex;
    atomicAdd(&segsum[seg[e]], ex);
}

__global__ void zero_kernel(float* __restrict__ p, size_t n) {
    size_t i = (size_t)blockIdx.x * WG + threadIdx.x;
    if (i < n) p[i] = 0.0f;
}

// aggr[tgt[e], d] += (ex[e]/segsum[seg[e]]) * msg[src[e], d]   (128 lanes/edge)
__global__ void aggregate_kernel(const float* __restrict__ ex,
                                 const float* __restrict__ segsum,
                                 const int* __restrict__ seg,
                                 const int* __restrict__ tgt,
                                 const int* __restrict__ src,
                                 const float* __restrict__ msg,
                                 float* __restrict__ aggr, int E_) {
    const size_t t = (size_t)blockIdx.x * WG + threadIdx.x;
    const int e = (int)(t >> 7);
    const int d = (int)(t & 127);
    if (e >= E_) return;
    const float coef = ex[e] / segsum[seg[e]];
    atomicAdd(&aggr[(size_t)tgt[e] * D + d], coef * msg[(size_t)src[e] * D + d]);
}

// ---------------------------------------------------------------------------
extern "C" void kernel_launch(void* const* d_in, const int* in_sizes, int n_in,
                              void* d_out, int out_size, void* d_ws, size_t ws_size,
                              hipStream_t stream) {
    const int*   l_edge  = (const int*)d_in[2];
    const int*   c_edge  = (const int*)d_in[3];
    const float* l_emb0  = (const float*)d_in[4];
    const float* c_emb0  = (const float*)d_in[5];
    const int E = in_sizes[2];
    const int L = in_sizes[4] / D;
    const int C = in_sizes[5] / D;

    // params flattened in dict insertion order
    const float* w1[3] = {(const float*)d_in[6],  (const float*)d_in[10], (const float*)d_in[14]};
    const float* b1[3] = {(const float*)d_in[7],  (const float*)d_in[11], (const float*)d_in[15]};
    const float* w2[3] = {(const float*)d_in[8],  (const float*)d_in[12], (const float*)d_in[16]};
    const float* b2[3] = {(const float*)d_in[9],  (const float*)d_in[13], (const float*)d_in[17]};
    const float* c_att  = (const float*)d_in[18];
    const float* l_att  = (const float*)d_in[19];
    const float* c_up_w = (const float*)d_in[20];
    const float* c_up_b = (const float*)d_in[21];
    const float* l_up_w = (const float*)d_in[22];
    const float* l_up_b = (const float*)d_in[23];

    // ---- workspace carve (256B aligned blocks) ----
    size_t off = 0;
    char* base = (char*)d_ws;
    auto carve = [&](size_t bytes) -> char* {
        char* p = base + off;
        off += (bytes + 255) & ~(size_t)255;
        return p;
    };
    bf16_t* wf[8];
    for (int i = 0; i < 6; ++i) wf[i] = (bf16_t*)carve((size_t)16384 * 2); // 128x128
    wf[6] = (bf16_t*)carve((size_t)32768 * 2);  // c_up: 256x128
    wf[7] = (bf16_t*)carve((size_t)49152 * 2);  // l_up: 384x128
    float*    l2c_out = (float*)carve((size_t)L * D * 4);
    float*    c2l_out = (float*)carve((size_t)C * D * 4);
    float*    l2l_out = (float*)carve((size_t)L * D * 4);
    float*    aggr_c  = (float*)carve((size_t)C * D * 4);
    float*    aggr_l  = (float*)carve((size_t)L * D * 4);
    float*    s_c   = (float*)carve((size_t)C * 4);
    float*    s_lm  = (float*)carve((size_t)L * 4);
    float*    s_l   = (float*)carve((size_t)L * 4);
    float*    s_cm  = (float*)carve((size_t)C * 4);
    unsigned* mA    = (unsigned*)carve((size_t)C * 4);
    float*    sumA  = (float*)carve((size_t)C * 4);
    unsigned* mB    = (unsigned*)carve((size_t)C * 4);
    float*    sumB  = (float*)carve((size_t)C * 4);
    float*    wA    = (float*)carve((size_t)E * 4);
    float*    wB    = (float*)carve((size_t)E * 4);

    // ---- one-time weight swizzle (bf16 B-fragment order) ----
    const float* wsrc[8] = {w1[0], w2[0], w1[1], w2[1], w1[2], w2[2], c_up_w, l_up_w};
    const int    ktl[8]  = {4, 4, 4, 4, 4, 4, 8, 12};
    for (int i = 0; i < 8; ++i)
        swizzle_w_kernel<<<(ktl[i] * 4096 + WG - 1) / WG, WG, 0, stream>>>(wsrc[i], ktl[i], wf[i]);

    // ---- output layout: l_embs [4,L,128] then c_embs [4,C,128] ----
    float* outL = (float*)d_out;
    float* outC = outL + (size_t)4 * L * D;
    hipMemcpyAsync(outL, l_emb0, (size_t)L * D * 4, hipMemcpyDeviceToDevice, stream);
    hipMemcpyAsync(outC, c_emb0, (size_t)C * D * 4, hipMemcpyDeviceToDevice, stream);

    const int gL = (L + 127) / 128, gC = (C + 127) / 128;
    const int gE = (E + WG - 1) / WG;

    for (int t = 0; t < 3; ++t) {
        const float* lE = outL + (size_t)t * L * D;
        const float* cE = outC + (size_t)t * C * D;
        float* lN = outL + (size_t)(t + 1) * L * D;
        float* cN = outC + (size_t)(t + 1) * C * D;

        // node-level MLPs (WMMA bf16)
        mlp2_fused_kernel<<<gL, WG, 0, stream>>>(lE, L, 0, wf[0], b1[0], wf[1], b2[0], l2c_out);
        mlp2_fused_kernel<<<gC, WG, 0, stream>>>(cE, C, 0, wf[2], b1[1], wf[3], b2[1], c2l_out);
        mlp2_fused_kernel<<<gL, WG, 0, stream>>>(lE, L, 1, wf[4], b1[2], wf[5], b2[2], l2l_out);

        // attention scalars: concat(...) @ att == two per-node dots
        nodedot_kernel<<<(C + 7) / 8, WG, 0, stream>>>(cE, C, c_att, s_c);
        nodedot_kernel<<<(L + 7) / 8, WG, 0, stream>>>(l2c_out, L, c_att + D, s_lm);
        nodedot_kernel<<<(L + 7) / 8, WG, 0, stream>>>(lE, L, l_att, s_l);
        nodedot_kernel<<<(C + 7) / 8, WG, 0, stream>>>(c2l_out, C, l_att + D, s_cm);

        // segment softmax over c_edge_index (both, faithful to reference)
        seginit_kernel<<<(C + WG - 1) / WG, WG, 0, stream>>>(mA, sumA, mB, sumB, C);
        edge_w_kernel<<<gE, WG, 0, stream>>>(c_edge, s_c, l_edge, s_lm, c_edge, mA, wA, E);
        edge_w_kernel<<<gE, WG, 0, stream>>>(l_edge, s_l, c_edge, s_cm, c_edge, mB, wB, E);
        edge_exp_kernel<<<gE, WG, 0, stream>>>(c_edge, mA, wA, sumA, E);
        edge_exp_kernel<<<gE, WG, 0, stream>>>(c_edge, mB, wB, sumB, E);

        // weighted scatter aggregation
        zero_kernel<<<(unsigned)(((size_t)C * D + WG - 1) / WG), WG, 0, stream>>>(aggr_c, (size_t)C * D);
        zero_kernel<<<(unsigned)(((size_t)L * D + WG - 1) / WG), WG, 0, stream>>>(aggr_l, (size_t)L * D);
        aggregate_kernel<<<(unsigned)(((size_t)E * D + WG - 1) / WG), WG, 0, stream>>>(
            wA, sumA, c_edge, c_edge, l_edge, l2c_out, aggr_c, E);
        aggregate_kernel<<<(unsigned)(((size_t)E * D + WG - 1) / WG), WG, 0, stream>>>(
            wB, sumB, c_edge, l_edge, c_edge, c2l_out, aggr_l, E);

        // update GEMMs (virtual concat, WMMA bf16)
        update_gemm_kernel<<<gC, WG, 0, stream>>>(cE, aggr_c, nullptr, C, wf[6], c_up_b, 8, cN);
        update_gemm_kernel<<<gL, WG, 0, stream>>>(lE, aggr_l, l2l_out, L, wf[7], l_up_b, 12, lN);
    }
}
